// RSSM_42460046688284
// MI455X (gfx1250) — matrix operand order repaired and checked by reading, not compiled, not run on hardware
//
#include <hip/hip_runtime.h>

typedef unsigned int   u32;
typedef unsigned short u16;
typedef __attribute__((ext_vector_type(16))) __bf16 v16bf;
typedef __attribute__((ext_vector_type(8)))  float  v8f;
typedef __attribute__((ext_vector_type(4)))  u32    u32x4;

#define DETER 1024
#define STOCH 32
#define NACT  6
#define HID   1024
#define NOBS  1024
#define BB    256
#define TT    64

__device__ __forceinline__ u16 f2bf(float f) {
  u32 x = __builtin_bit_cast(u32, f);
  x += 0x7FFFu + ((x >> 16) & 1u);          // round-to-nearest-even
  return (u16)(x >> 16);
}
__device__ __forceinline__ float bf2f(u16 h) {
  u32 x = ((u32)h) << 16;
  return __builtin_bit_cast(float, x);
}

// ---------------------------------------------------------------------------
// Dual-job bf16 WMMA GEMM:  C[M,N] = act( A[M,K] @ W[N,K]^T + bias[N] )
// blockIdx.z picks the job -> two independent GEMMs fused into one launch.
// Workgroup tile 64x64, BK=32, 128 threads = 4 waves in 2(M) x 2(N) grid,
// each wave computes a 32x32 block as 2x2 v_wmma_f32_16x16x32_bf16 tiles.
// LDS is double-buffered: one __syncthreads per K-chunk; global loads for
// chunk k+1 overlap the WMMAs of chunk k.
// ---------------------------------------------------------------------------
struct GemmJob {
  const u16* A;          // [M,K] bf16 row-major
  const u16* W;          // [N,K] bf16 row-major
  const float* bias;     // [N]
  float* Cf;             // fp32 out (or null)
  u16*   Cb;             // bf16 out (or null)
  int N, K, relu;
};

__device__ __forceinline__ void store_tile(v8f cc, int m_base, int n, int hi8,
                                           float bval, int relu,
                                           float* Cf, u16* Cb, int N) {
#pragma unroll
  for (int r = 0; r < 8; ++r) {
    int m = m_base + r + hi8;
    float v = cc[r] + bval;
    if (relu) v = fmaxf(v, 0.0f);
    size_t idx = (size_t)m * N + n;
    if (Cf) Cf[idx] = v;
    if (Cb) Cb[idx] = f2bf(v);
  }
}

__global__ __launch_bounds__(128)
void gemm_bf16(GemmJob j0, GemmJob j1) {
  const GemmJob j = (blockIdx.z == 0) ? j0 : j1;
  __shared__ __align__(16) u16 lA[2][64][40];   // double-buffered 64x32 tiles
  __shared__ __align__(16) u16 lB[2][64][40];

  const int tid  = threadIdx.x;
  const int lane = tid & 31;
  const int wave = tid >> 5;
  const int wm   = wave >> 1;          // 0..1 -> M offset wm*32
  const int wn   = wave & 1;           // 0..1 -> N offset wn*32
  const int lm   = lane & 15;
  const int hi   = lane >> 4;          // lane half
  const int aoff = hi ? 8 : 0;         // A-frag K offset per ISA layout
  const int boff = hi ? 8 : 0;         // B-frag K offset per ISA layout

  const int m0 = blockIdx.y * 64;
  const int n0 = blockIdx.x * 64;
  const int K = j.K, N = j.N;
  const u16* __restrict__ A = j.A;
  const u16* __restrict__ W = j.W;

  v8f c00 = {}, c01 = {}, c10 = {}, c11 = {};

  auto stage = [&](int k0, int p) {
#pragma unroll
    for (int i = 0; i < 2; ++i) {
      int id = tid + i * 128;
      int r  = id >> 2;
      int cb = (id & 3) << 3;
      *(u32x4*)&lA[p][r][cb] = *(const u32x4*)(A + (size_t)(m0 + r) * K + k0 + cb);
    }
#pragma unroll
    for (int i = 0; i < 2; ++i) {
      int id = tid + i * 128;
      int r  = id >> 2;
      int cb = (id & 3) << 3;
      *(u32x4*)&lB[p][r][cb] = *(const u32x4*)(W + (size_t)(n0 + r) * K + k0 + cb);
    }
  };

  auto compute = [&](int p) {
    union Frag { v16bf v; u32x4 q[2]; } a0, a1, b0, b1;
    int mA0 = wm * 32 + lm;
    int mA1 = mA0 + 16;
    a0.q[0] = *(const u32x4*)&lA[p][mA0][aoff];
    a0.q[1] = *(const u32x4*)&lA[p][mA0][aoff + 16];
    a1.q[0] = *(const u32x4*)&lA[p][mA1][aoff];
    a1.q[1] = *(const u32x4*)&lA[p][mA1][aoff + 16];
    int nB0 = wn * 32 + lm;
    int nB1 = nB0 + 16;
    b0.q[0] = *(const u32x4*)&lB[p][nB0][boff];
    b0.q[1] = *(const u32x4*)&lB[p][nB0][boff + 16];
    b1.q[0] = *(const u32x4*)&lB[p][nB1][boff];
    b1.q[1] = *(const u32x4*)&lB[p][nB1][boff + 16];
    c00 = __builtin_amdgcn_wmma_f32_16x16x32_bf16(false, a0.v, false, b0.v, (short)0, c00, false, false);
    c01 = __builtin_amdgcn_wmma_f32_16x16x32_bf16(false, a0.v, false, b1.v, (short)0, c01, false, false);
    c10 = __builtin_amdgcn_wmma_f32_16x16x32_bf16(false, a1.v, false, b0.v, (short)0, c10, false, false);
    c11 = __builtin_amdgcn_wmma_f32_16x16x32_bf16(false, a1.v, false, b1.v, (short)0, c11, false, false);
  };

  stage(0, 0);
  __syncthreads();
  int p = 0;
  for (int k0 = 32; k0 < K; k0 += 32) {
    stage(k0, p ^ 1);     // prefetch next chunk into the other buffer
    compute(p);           // WMMAs on current buffer overlap the loads
    __syncthreads();
    p ^= 1;
  }
  compute(p);

  const int hi8 = hi * 8;
  {
    int n = n0 + wn * 32 + lm;
    float bv = j.bias ? j.bias[n] : 0.0f;
    store_tile(c00, m0 + wm * 32,      n, hi8, bv, j.relu, j.Cf, j.Cb, N);
    store_tile(c10, m0 + wm * 32 + 16, n, hi8, bv, j.relu, j.Cf, j.Cb, N);
  }
  {
    int n = n0 + wn * 32 + 16 + lm;
    float bv = j.bias ? j.bias[n] : 0.0f;
    store_tile(c01, m0 + wm * 32,      n, hi8, bv, j.relu, j.Cf, j.Cb, N);
    store_tile(c11, m0 + wm * 32 + 16, n, hi8, bv, j.relu, j.Cf, j.Cb, N);
  }
}

// ---------------------------------------------------------------------------
// Small VALU kernels for the cheap stages
// ---------------------------------------------------------------------------
__global__ void cvt_bf16(const float* __restrict__ src, u16* __restrict__ dst, int n) {
  int i = blockIdx.x * 256 + threadIdx.x;
  if (i < n) dst[i] = f2bf(src[i]);
}

__global__ void init_state(float* h, u16* hb, float* z) {
  int i = blockIdx.x * 256 + threadIdx.x;
  if (i < BB * DETER) { h[i] = 0.0f; hb[i] = 0; }
  if (i < BB * STOCH) z[i] = 0.0f;
}

// x1 = relu([z, prev_a] @ in_w1^T + b1)   (K=38, trivial -> plain fp32)
__global__ __launch_bounds__(256)
void input_layer1(const float* __restrict__ z, const float* __restrict__ actions, int t,
                  const float* __restrict__ w1, const float* __restrict__ b1,
                  u16* __restrict__ x1) {
  int idx = blockIdx.x * 256 + threadIdx.x;      // BB*HID
  int b = idx >> 10, o = idx & 1023;
  const float* wr = w1 + (size_t)o * (STOCH + NACT);
  const float* zb = z + (size_t)b * STOCH;
  float acc = b1[o];
#pragma unroll
  for (int k = 0; k < STOCH; ++k) acc += zb[k] * wr[k];
  if (t > 0) {
    const float* ab = actions + ((size_t)b * TT + (t - 1)) * NACT;
#pragma unroll
    for (int k = 0; k < NACT; ++k) acc += ab[k] * wr[STOCH + k];
  }
  x1[idx] = f2bf(fmaxf(acc, 0.0f));
}

// GRU gates + h update; also emits h(bf16), posterior concat [h|obs](bf16),
// and the h-part of feat directly into d_out.
__global__ __launch_bounds__(256)
void gru_gate(const float* __restrict__ gi, const float* __restrict__ gh,
              float* __restrict__ h, u16* __restrict__ h_bf,
              const float* __restrict__ obs, int t,
              u16* __restrict__ qcat, float* __restrict__ feat) {
  int idx = blockIdx.x * 256 + threadIdx.x;      // BB*DETER
  int b = idx >> 10, i = idx & 1023;
  size_t gb = (size_t)b * 3 * DETER;
  float ir = gi[gb + i], iz = gi[gb + DETER + i], in = gi[gb + 2 * DETER + i];
  float hr = gh[gb + i], hz = gh[gb + DETER + i], hn = gh[gb + 2 * DETER + i];
  float r = 1.0f / (1.0f + expf(-(ir + hr)));
  float u = 1.0f / (1.0f + expf(-(iz + hz)));
  float n = tanhf(in + r * hn);
  float hnew = (1.0f - u) * n + u * h[idx];
  h[idx] = hnew;
  u16 hb = f2bf(hnew);
  h_bf[idx] = hb;
  qcat[(size_t)b * 2048 + i] = hb;
  qcat[(size_t)b * 2048 + 1024 + i] = f2bf(obs[((size_t)b * TT + t) * NOBS + i]);
  feat[((size_t)b * TT + t) * 1056 + i] = hnew;
}

// All four 32-wide head projections + reparameterized z + feat tail, fused.
// branch 0: prior (pm, ps);  branch 1: posterior (qm, qs, z, feat[1024:]).
__global__ __launch_bounds__(256)
void heads_fused(const u16* __restrict__ px2, const u16* __restrict__ qx2,
                 const u16* __restrict__ wpm, const float* __restrict__ bpm,
                 const u16* __restrict__ wps, const float* __restrict__ bps,
                 const u16* __restrict__ wqm, const float* __restrict__ bqm,
                 const u16* __restrict__ wqs, const float* __restrict__ bqs,
                 const float* __restrict__ eps_post, int t,
                 float* __restrict__ o_pm, float* __restrict__ o_ps,
                 float* __restrict__ o_qm, float* __restrict__ o_qs,
                 float* __restrict__ zf, float* __restrict__ feat) {
  int idx = blockIdx.x * 256 + threadIdx.x;      // 2*BB*STOCH = 16384
  int branch = idx >> 13;
  int b = (idx >> 5) & (BB - 1);
  int jj = idx & 31;
  const u16* ar = (branch ? qx2 : px2) + (size_t)b * HID;
  const u16* wmr = (branch ? wqm : wpm) + (size_t)jj * HID;
  const u16* wsr = (branch ? wqs : wps) + (size_t)jj * HID;
  float am = (branch ? bqm : bpm)[jj];
  float as = (branch ? bqs : bps)[jj];
  for (int k = 0; k < HID; ++k) {
    float a = bf2f(ar[k]);
    am += a * bf2f(wmr[k]);
    as += a * bf2f(wsr[k]);
  }
  as = expf(fminf(fmaxf(as, -5.0f), 2.0f));
  size_t o = ((size_t)b * TT + t) * STOCH + jj;
  if (branch == 0) {
    o_pm[o] = am;
    o_ps[o] = as;
  } else {
    o_qm[o] = am;
    o_qs[o] = as;
    float zv = am + as * eps_post[o];
    zf[b * STOCH + jj] = zv;
    feat[((size_t)b * TT + t) * 1056 + DETER + jj] = zv;
  }
}

// ---------------------------------------------------------------------------
extern "C" void kernel_launch(void* const* d_in, const int* in_sizes, int n_in,
                              void* d_out, int out_size, void* d_ws, size_t ws_size,
                              hipStream_t stream) {
  (void)in_sizes; (void)n_in; (void)out_size; (void)ws_size;

  const float* obs      = (const float*)d_in[0];
  const float* actions  = (const float*)d_in[1];
  /* d_in[2] = eps_prior: unused by the reference outputs */
  const float* eps_post = (const float*)d_in[3];
  const float* in_w1 = (const float*)d_in[4];
  const float* in_b1 = (const float*)d_in[5];
  const float* in_w2 = (const float*)d_in[6];
  const float* in_b2 = (const float*)d_in[7];
  const float* gru_wih = (const float*)d_in[8];
  const float* gru_whh = (const float*)d_in[9];
  const float* gru_bih = (const float*)d_in[10];
  const float* gru_bhh = (const float*)d_in[11];
  const float* pr_w1 = (const float*)d_in[12];
  const float* pr_b1 = (const float*)d_in[13];
  const float* pr_w2 = (const float*)d_in[14];
  const float* pr_b2 = (const float*)d_in[15];
  const float* pr_mw = (const float*)d_in[16];
  const float* pr_mb = (const float*)d_in[17];
  const float* pr_sw = (const float*)d_in[18];
  const float* pr_sb = (const float*)d_in[19];
  const float* po_w1 = (const float*)d_in[20];
  const float* po_b1 = (const float*)d_in[21];
  const float* po_w2 = (const float*)d_in[22];
  const float* po_b2 = (const float*)d_in[23];
  const float* po_mw = (const float*)d_in[24];
  const float* po_mb = (const float*)d_in[25];
  const float* po_sw = (const float*)d_in[26];
  const float* po_sb = (const float*)d_in[27];

  char* ws = (char*)d_ws;
  size_t off = 0;
  auto take = [&](size_t bytes) -> void* {
    void* p = ws + off;
    off = (off + bytes + 255) & ~(size_t)255;
    return p;
  };
  // bf16 weight copies (L2-resident across the whole rollout: ~25 MB)
  u16* wb_in2 = (u16*)take((size_t)HID * HID * 2);
  u16* wb_wih = (u16*)take((size_t)3 * DETER * HID * 2);
  u16* wb_whh = (u16*)take((size_t)3 * DETER * DETER * 2);
  u16* wb_pr1 = (u16*)take((size_t)HID * DETER * 2);
  u16* wb_pr2 = (u16*)take((size_t)HID * HID * 2);
  u16* wb_po1 = (u16*)take((size_t)HID * (DETER + NOBS) * 2);
  u16* wb_po2 = (u16*)take((size_t)HID * HID * 2);
  u16* wb_prm = (u16*)take((size_t)STOCH * HID * 2);
  u16* wb_prs = (u16*)take((size_t)STOCH * HID * 2);
  u16* wb_pom = (u16*)take((size_t)STOCH * HID * 2);
  u16* wb_pos = (u16*)take((size_t)STOCH * HID * 2);
  // per-step activations
  u16* x1   = (u16*)take((size_t)BB * HID * 2);
  u16* x2   = (u16*)take((size_t)BB * HID * 2);
  u16* hbf  = (u16*)take((size_t)BB * DETER * 2);
  u16* qcat = (u16*)take((size_t)BB * (DETER + NOBS) * 2);
  u16* px1  = (u16*)take((size_t)BB * HID * 2);
  u16* px2  = (u16*)take((size_t)BB * HID * 2);
  u16* qx1  = (u16*)take((size_t)BB * HID * 2);
  u16* qx2  = (u16*)take((size_t)BB * HID * 2);
  float* gi = (float*)take((size_t)BB * 3 * DETER * 4);
  float* gh = (float*)take((size_t)BB * 3 * DETER * 4);
  float* hf = (float*)take((size_t)BB * DETER * 4);
  float* zf = (float*)take((size_t)BB * STOCH * 4);

  float* out = (float*)d_out;
  const size_t BT = (size_t)BB * TT;
  float* o_feat = out;
  float* o_pm = out + BT * 1056;
  float* o_ps = o_pm + BT * STOCH;
  float* o_qm = o_ps + BT * STOCH;
  float* o_qs = o_qm + BT * STOCH;

  auto cvt = [&](const float* s, u16* d, size_t n) {
    cvt_bf16<<<(unsigned)((n + 255) / 256), 256, 0, stream>>>(s, d, (int)n);
  };
  cvt(in_w2,   wb_in2, (size_t)HID * HID);
  cvt(gru_wih, wb_wih, (size_t)3 * DETER * HID);
  cvt(gru_whh, wb_whh, (size_t)3 * DETER * DETER);
  cvt(pr_w1,   wb_pr1, (size_t)HID * DETER);
  cvt(pr_w2,   wb_pr2, (size_t)HID * HID);
  cvt(po_w1,   wb_po1, (size_t)HID * (DETER + NOBS));
  cvt(po_w2,   wb_po2, (size_t)HID * HID);
  cvt(pr_mw,   wb_prm, (size_t)STOCH * HID);
  cvt(pr_sw,   wb_prs, (size_t)STOCH * HID);
  cvt(po_mw,   wb_pom, (size_t)STOCH * HID);
  cvt(po_sw,   wb_pos, (size_t)STOCH * HID);
  init_state<<<BB * DETER / 256, 256, 0, stream>>>(hf, hbf, zf);

  dim3 blk128(128), blk256(256);
  dim3 gX2 (1024 / 64, 256 / 64, 1);   // single job, N=1024
  dim3 gGRU(3072 / 64, 256 / 64, 2);   // dual job, N=3072 (gi & gh)
  dim3 gL  (1024 / 64, 256 / 64, 2);   // dual job, N=1024 (prior & posterior)

  for (int t = 0; t < TT; ++t) {
    input_layer1<<<BB * HID / 256, blk256, 0, stream>>>(zf, actions, t, in_w1, in_b1, x1);

    GemmJob jx2 = { x1, wb_in2, in_b2, nullptr, x2, 1024, 1024, 1 };
    gemm_bf16<<<gX2, blk128, 0, stream>>>(jx2, jx2);

    GemmJob jgi = { x2,  wb_wih, gru_bih, gi, nullptr, 3072, 1024, 0 };
    GemmJob jgh = { hbf, wb_whh, gru_bhh, gh, nullptr, 3072, 1024, 0 };
    gemm_bf16<<<gGRU, blk128, 0, stream>>>(jgi, jgh);

    gru_gate<<<BB * DETER / 256, blk256, 0, stream>>>(gi, gh, hf, hbf, obs, t, qcat, o_feat);

    GemmJob jp1 = { hbf,  wb_pr1, pr_b1, nullptr, px1, 1024, 1024, 1 };
    GemmJob jq1 = { qcat, wb_po1, po_b1, nullptr, qx1, 1024, 2048, 1 };
    gemm_bf16<<<gL, blk128, 0, stream>>>(jp1, jq1);

    GemmJob jp2 = { px1, wb_pr2, pr_b2, nullptr, px2, 1024, 1024, 1 };
    GemmJob jq2 = { qx1, wb_po2, po_b2, nullptr, qx2, 1024, 1024, 1 };
    gemm_bf16<<<gL, blk128, 0, stream>>>(jp2, jq2);

    heads_fused<<<2 * BB * STOCH / 256, blk256, 0, stream>>>(
        px2, qx2, wb_prm, pr_mb, wb_prs, pr_sb, wb_pom, po_mb, wb_pos, po_sb,
        eps_post, t, o_pm, o_ps, o_qm, o_qs, zf, o_feat);
  }
}
